// Attention_30468497997979
// MI455X (gfx1250) — matrix operand order repaired and checked by reading, not compiled, not run on hardware
//
#include <hip/hip_runtime.h>

typedef __attribute__((ext_vector_type(16))) _Float16 v16h;
typedef __attribute__((ext_vector_type(8)))  _Float16 v8h;
typedef __attribute__((ext_vector_type(4)))  _Float16 v4h;
typedef __attribute__((ext_vector_type(8)))  float    v8f;
typedef __attribute__((ext_vector_type(4)))  float    v4f;

#define B_ 8
#define C_ 768
#define S_ 384
#define H_ 12
#define D_ 64
#define CKS 24   // C_/32 reduction chunks
#define OTS 12   // C_/64 output-row tiles
#define SBLKS 6  // S_/64 spatial blocks

static __device__ __forceinline__ v8f wmma_f16(v16h a, v16h b, v8f c) {
  // D(16x16 f32) = A(16x32 f16) * B(32x16 f16) + C
  return __builtin_amdgcn_wmma_f32_16x16x32_f16(false, a, false, b, (short)0, c,
                                                false, false);
}

// CDNA5 async global->LDS copy (tracked with ASYNCcnt). 16B per lane.
static __device__ __forceinline__ void async_copy_b128(unsigned lds_off,
                                                       const void* gaddr) {
  asm volatile("global_load_async_to_lds_b128 %0, %1, off"
               :: "v"(lds_off), "v"(gaddr) : "memory");
}
static __device__ __forceinline__ void wait_async0() {
  asm volatile("s_wait_asynccnt 0x0" ::: "memory");
}

// ---------------------------------------------------------------------------
// Kernel 0: one-shot fp32->f16 weight conversion, permuted into WMMA
// A-fragment order:  wperm[w][ot][ck][tl][lane*16+elem]
//   (w=0..3: Wq,Wk,Wv,Wo;  ot=o/64;  ck=c/32;  tl=(o%64)/16)
// A 4-aligned column run stays inside one interleave group -> one float4 load
// and one contiguous 8B f16 store per thread.
// ---------------------------------------------------------------------------
__global__ __launch_bounds__(256)
void permute_weights(const float* __restrict__ Wq, const float* __restrict__ Wk,
                     const float* __restrict__ Wv, const float* __restrict__ Wo,
                     _Float16* __restrict__ wperm)
{
  const int gid = blockIdx.x * 256 + threadIdx.x;     // 4 * 768 * 192 total
  const int w   = gid / (C_ * (C_ / 4));
  const int rem = gid % (C_ * (C_ / 4));
  const int o   = rem / (C_ / 4);
  const int c0  = (rem % (C_ / 4)) * 4;

  const float* W = (w == 0) ? Wq : (w == 1) ? Wk : (w == 2) ? Wv : Wo;
  v4f x = *(const v4f*)&W[(size_t)o * C_ + c0];

  const int ot = o >> 6, ml = o & 63, tl = ml >> 4, m = ml & 15;
  const int ck = c0 >> 5, cl = c0 & 31;
  const int L  = ((cl & 15) >> 3) * 16 + m;
  const int e  = (cl >> 4) * 8 + (cl & 7);            // 4-aligned -> run of 4

  v4h hv;
#pragma unroll
  for (int j = 0; j < 4; ++j) hv[j] = (_Float16)x[j];
  size_t dst = ((((size_t)w * OTS + ot) * CKS + ck) * 4 + tl) * 512 + L * 16 + e;
  *(v4h*)&wperm[dst] = hv;
}

// ---------------------------------------------------------------------------
// Kernel 1: fused Q/K/V projection.  grid = (12, 6, B*3), block = 128.
// A tiles: async bulk copy of pre-permuted f16 weights (4 KB / iter).
// B tiles: fp32 X gathered 4-along-c per thread -> one ds_store_b64.
// ---------------------------------------------------------------------------
__global__ __launch_bounds__(128)
void proj_qkv(const float* __restrict__ X, const _Float16* __restrict__ wperm,
              const float* __restrict__ bq, const float* __restrict__ bk,
              const float* __restrict__ bv,
              _Float16* __restrict__ qws, _Float16* __restrict__ kws,
              _Float16* __restrict__ vws)
{
  __shared__ __align__(16) _Float16 lds_a[2048];
  __shared__ __align__(16) _Float16 lds_b[2048];

  const int tid  = threadIdx.x;
  const int lane = tid & 31;
  const int wave = tid >> 5;
  const int hlf  = lane >> 4;

  const int ob = blockIdx.x * 64;
  const int sb = blockIdx.y * 64;
  const int b  = blockIdx.z / 3;
  const int p  = blockIdx.z % 3;

  const float* bias = (p == 0) ? bq : (p == 1) ? bk : bv;
  const float* Xb   = X + (size_t)b * C_ * S_;
  const _Float16* wbase = wperm + ((size_t)p * OTS + blockIdx.x) * CKS * 2048;

  const unsigned la = (unsigned)(size_t)&lds_a[0];

  v8f zero = {};
  v8f acc[4];
#pragma unroll
  for (int t = 0; t < 4; ++t) acc[t] = zero;

  for (int ck = 0; ck < CKS; ++ck) {
    // A tile: async 4KB copy, 32B per thread
    const char* ga = (const char*)(wbase + (size_t)ck * 2048) + tid * 32;
    async_copy_b128(la + tid * 32, ga);
    async_copy_b128(la + tid * 32 + 16, ga + 16);

    // B tile: 32c x 64s fp32 -> f16, fragment order. 4 groups of 4-along-c.
    float xs[16];
#pragma unroll
    for (int g = 0; g < 4; ++g) {
      int gi = g * 128 + tid;                 // 0..511
      int c0 = (gi >> 6) * 4, s = gi & 63;
#pragma unroll
      for (int j = 0; j < 4; ++j)
        xs[g * 4 + j] = Xb[(size_t)(ck * 32 + c0 + j) * S_ + sb + s];
    }
#pragma unroll
    for (int g = 0; g < 4; ++g) {
      int gi = g * 128 + tid;
      int c0 = (gi >> 6) * 4, s = gi & 63;
      v4h hv;
#pragma unroll
      for (int j = 0; j < 4; ++j) hv[j] = (_Float16)xs[g * 4 + j];
      int L = ((c0 & 15) >> 3) * 16 + (s & 15);
      *(v4h*)&lds_b[(s >> 4) * 512 + L * 16 + (c0 >> 4) * 8 + (c0 & 7)] = hv;
    }
    wait_async0();
    __syncthreads();

    v16h a = *(const v16h*)&lds_a[wave * 512 + lane * 16];
#pragma unroll
    for (int t = 0; t < 4; ++t) {
      v16h bf = *(const v16h*)&lds_b[t * 512 + lane * 16];
      acc[t] = wmma_f16(a, bf, acc[t]);
    }
    __syncthreads();
  }

  // epilogue: bias, scatter to attention layouts (Q pre-scaled by D^-1/2)
#pragma unroll
  for (int t = 0; t < 4; ++t) {
    int s = sb + t * 16 + (lane & 15);
#pragma unroll
    for (int r = 0; r < 8; ++r) {
      int o   = ob + wave * 16 + r + 8 * hlf;      // D-layout m = r + 8*(lane/16)
      float v = acc[t][r] + bias[o];
      int h = o % H_, d = o / H_;                  // torch reshape: c = d*H + h
      if (p == 0)
        qws[(((size_t)b * H_ + h) * S_ + s) * D_ + d] = (_Float16)(v * 0.125f);
      else if (p == 1)
        kws[(((size_t)b * H_ + h) * S_ + s) * D_ + d] = (_Float16)v;
      else // V transposed [d][s] so PV B-fragments are lane-contiguous
        vws[(((size_t)b * H_ + h) * D_ + d) * S_ + s] = (_Float16)v;
    }
  }
}

// ---------------------------------------------------------------------------
// Kernel 2: flash attention per (b,h).  grid = (6, 12, 8), block = 128.
// Output written directly in out_proj's B-fragment order (aperm).
// ---------------------------------------------------------------------------
__global__ __launch_bounds__(128)
void attn_kernel(const _Float16* __restrict__ qws, const _Float16* __restrict__ kws,
                 const _Float16* __restrict__ vws, _Float16* __restrict__ aperm)
{
  const int tid  = threadIdx.x;
  const int lane = tid & 31;
  const int wave = tid >> 5;
  const int hlf  = lane >> 4;
  const int col  = lane & 15;

  const int h  = blockIdx.y;
  const int b  = blockIdx.z;
  const int q0 = blockIdx.x * 64 + wave * 16;

  const _Float16* Q = qws + ((size_t)b * H_ + h) * S_ * D_;   // [s][d]
  const _Float16* K = kws + ((size_t)b * H_ + h) * S_ * D_;   // [s][d]
  const _Float16* V = vws + ((size_t)b * H_ + h) * D_ * S_;   // [d][s]

  v16h qf[2];                    // persistent Q B-fragments (k-dim = d)
#pragma unroll
  for (int dc = 0; dc < 2; ++dc)
    qf[dc] = *(const v16h*)&Q[(size_t)(q0 + col) * D_ + dc * 32 + hlf * 16];

  v8f zero = {};
  v8f o_acc[4];
#pragma unroll
  for (int t = 0; t < 4; ++t) o_acc[t] = zero;
  float mrun = -1e30f, lrun = 0.0f;
  const int qg = q0 + col;

  for (int kb = 0; kb < S_; kb += 32) {
    v8f sc[2];
    sc[0] = zero; sc[1] = zero;
#pragma unroll
    for (int dc = 0; dc < 2; ++dc) {
#pragma unroll
      for (int t = 0; t < 2; ++t) {
        union { v16h v; v8h h8[2]; } af;   // K A-frag: two contiguous 16B runs
        const _Float16* krow =
            &K[(size_t)(kb + t * 16 + col) * D_ + dc * 32 + hlf * 16];
        af.h8[0] = *(const v8h*)krow;
        af.h8[1] = *(const v8h*)(krow + 8);
        sc[t] = wmma_f16(af.v, qf[dc], sc[t]);
      }
    }
    // mask (-1e4 where q > k) + online softmax over keys
    float pvals[16];
    float lmax = -1e30f;
#pragma unroll
    for (int t = 0; t < 2; ++t)
#pragma unroll
      for (int r = 0; r < 8; ++r) {
        int kg  = kb + t * 16 + r + 8 * hlf;
        float s = sc[t][r] + ((qg > kg) ? -10000.0f : 0.0f);
        pvals[t * 8 + r] = s;
        lmax = fmaxf(lmax, s);
      }
    lmax = fmaxf(lmax, __shfl_xor(lmax, 16, 32));
    float mnew  = fmaxf(mrun, lmax);
    float alpha = __expf(mrun - mnew);
    mrun = mnew;
    lrun *= alpha;
#pragma unroll
    for (int t = 0; t < 4; ++t)
#pragma unroll
      for (int r = 0; r < 8; ++r) o_acc[t][r] *= alpha;

    // P in A-fragment order: lane-local mapping, no shuffles/LDS
    union { v16h v; _Float16 hh[16]; } pf;
    float psum = 0.0f;
#pragma unroll
    for (int i = 0; i < 16; ++i) {
      float pv = __expf(pvals[i] - mnew);
      psum += pv;
      pf.hh[i] = (_Float16)pv;
    }
    psum += __shfl_xor(psum, 16, 32);
    lrun += psum;

#pragma unroll
    for (int t = 0; t < 4; ++t) {
      union { v16h v; v8h h8[2]; } vf;
      const _Float16* vrow = &V[(size_t)(t * 16 + col) * S_ + kb + hlf * 8];
      vf.h8[0] = *(const v8h*)vrow;
      vf.h8[1] = *(const v8h*)(vrow + 16);
      o_acc[t] = wmma_f16(pf.v, vf.v, o_acc[t]);
    }
  }

  float inv = 1.0f / lrun;
  // store into aperm[b][ck][sblk][nt][L*16+e]  (out_proj B-fragment order)
  //   c = h*64 + t*16 + col -> ck = h*2 + (t>>1), e = (t&1)*8 + (col&7), L>>4 = col>>3
  //   s = q0 + r + 8*hlf    -> sblk = bx, nt = wave, n = r + 8*hlf
#pragma unroll
  for (int t = 0; t < 4; ++t) {
    size_t base =
        ((((size_t)b * CKS + h * 2 + (t >> 1)) * SBLKS + blockIdx.x) * 4 + wave)
        * 512;
#pragma unroll
    for (int r = 0; r < 8; ++r)
      aperm[base + (size_t)(((col >> 3) * 16 + r + 8 * hlf) * 16 + (t & 1) * 8 + (col & 7))] =
          (_Float16)(o_acc[t][r] * inv);
  }
}

// ---------------------------------------------------------------------------
// Kernel 3: output projection.  grid = (12, 6, 8), block = 128.
// Both operands are pre-permuted f16 -> pure async bulk copies into LDS.
// ---------------------------------------------------------------------------
__global__ __launch_bounds__(128)
void out_proj(const _Float16* __restrict__ aperm, const _Float16* __restrict__ wperm,
              float* __restrict__ out)
{
  __shared__ __align__(16) _Float16 lds_a[2048];
  __shared__ __align__(16) _Float16 lds_b[2048];

  const int tid  = threadIdx.x;
  const int lane = tid & 31;
  const int wave = tid >> 5;
  const int hlf  = lane >> 4;
  const int ob = blockIdx.x * 64;
  const int sb = blockIdx.y * 64;
  const int b  = blockIdx.z;

  const _Float16* wbase = wperm + ((size_t)3 * OTS + blockIdx.x) * CKS * 2048;
  const _Float16* abase = aperm + ((size_t)b * CKS * SBLKS + blockIdx.y) * 2048;

  const unsigned la = (unsigned)(size_t)&lds_a[0];
  const unsigned lb = (unsigned)(size_t)&lds_b[0];

  v8f zero = {};
  v8f acc[4];
#pragma unroll
  for (int t = 0; t < 4; ++t) acc[t] = zero;

  for (int ck = 0; ck < CKS; ++ck) {
    const char* ga = (const char*)(wbase + (size_t)ck * 2048) + tid * 32;
    const char* gb = (const char*)(abase + (size_t)ck * SBLKS * 2048) + tid * 32;
    async_copy_b128(la + tid * 32,      ga);
    async_copy_b128(la + tid * 32 + 16, ga + 16);
    async_copy_b128(lb + tid * 32,      gb);
    async_copy_b128(lb + tid * 32 + 16, gb + 16);
    wait_async0();
    __syncthreads();

    v16h a = *(const v16h*)&lds_a[wave * 512 + lane * 16];
#pragma unroll
    for (int t = 0; t < 4; ++t) {
      v16h bf = *(const v16h*)&lds_b[t * 512 + lane * 16];
      acc[t] = wmma_f16(a, bf, acc[t]);
    }
    __syncthreads();
  }

#pragma unroll
  for (int t = 0; t < 4; ++t) {
    int s = sb + t * 16 + (lane & 15);
#pragma unroll
    for (int r = 0; r < 8; ++r) {
      int o = ob + wave * 16 + r + 8 * hlf;
      out[((size_t)b * C_ + o) * S_ + s] = acc[t][r];
    }
  }
}

// ---------------------------------------------------------------------------
extern "C" void kernel_launch(void* const* d_in, const int* in_sizes, int n_in,
                              void* d_out, int out_size, void* d_ws, size_t ws_size,
                              hipStream_t stream) {
  (void)in_sizes; (void)n_in; (void)out_size; (void)ws_size;
  const float* X  = (const float*)d_in[0];
  const float* Wq = (const float*)d_in[1];
  const float* bq = (const float*)d_in[2];
  const float* Wk = (const float*)d_in[3];
  const float* bk = (const float*)d_in[4];
  const float* Wv = (const float*)d_in[5];
  const float* bv = (const float*)d_in[6];
  const float* Wo = (const float*)d_in[7];
  float* out = (float*)d_out;

  const size_t nW = (size_t)4 * C_ * C_;          // 2,359,296 f16
  const size_t nQ = (size_t)B_ * H_ * S_ * D_;    // 2,359,296 f16
  _Float16* wperm = (_Float16*)d_ws;
  _Float16* qws   = wperm + nW;
  _Float16* kws   = qws + nQ;
  _Float16* vws   = kws + nQ;
  _Float16* aperm = vws + nQ;                     // total ~23.6 MB of workspace

  permute_weights<<<dim3(4 * C_ * (C_ / 4) / 256), 256, 0, stream>>>(
      Wq, Wk, Wv, Wo, wperm);
  proj_qkv<<<dim3(OTS, SBLKS, B_ * 3), 128, 0, stream>>>(
      X, wperm, bq, bk, bv, qws, kws, vws);
  attn_kernel<<<dim3(SBLKS, H_, B_), 128, 0, stream>>>(qws, kws, vws, aperm);
  out_proj<<<dim3(OTS, SBLKS, B_), 128, 0, stream>>>(aperm, wperm, out);
}
